// MoE_10514079941231
// MI455X (gfx1250) — matrix operand order repaired and checked by reading, not compiled, not run on hardware
//
#include <hip/hip_runtime.h>
#include <math.h>

typedef __attribute__((ext_vector_type(16))) __bf16 v16bf;
typedef __attribute__((ext_vector_type(8)))  float  v8f;

#define DDIM 1024
#define NE   8
#define BM   128
#define BN   128
#define KS   128          // K elements staged in LDS per stage
#define AROW 136          // bf16 elems per LDS A row (272B, 16B-aligned, bank-diffused)
#define CROW 40           // bf16 elems per comb/beT row (80B)

// ---------------------------------------------------------------------------
// Gating: one wave (32 lanes) per token. Computes dense combine[t][8] =
// softmax over the strict top-2 of (x @ Wg^T + bg), zeros elsewhere.
// ---------------------------------------------------------------------------
__global__ __launch_bounds__(256) void gate_kernel(
    const float* __restrict__ x, const float* __restrict__ Wg,
    const float* __restrict__ bg, float* __restrict__ comb) {
  const int wid  = threadIdx.x >> 5;
  const int lane = threadIdx.x & 31;
  const int t    = blockIdx.x * 8 + wid;

  float acc[NE];
#pragma unroll
  for (int e = 0; e < NE; ++e) acc[e] = 0.f;

  const float* xr = x + (size_t)t * DDIM;
  for (int d = lane; d < DDIM; d += 32) {
    float xv = xr[d];
#pragma unroll
    for (int e = 0; e < NE; ++e) acc[e] += xv * Wg[e * DDIM + d];
  }
#pragma unroll
  for (int e = 0; e < NE; ++e) {
#pragma unroll
    for (int off = 16; off > 0; off >>= 1)
      acc[e] += __shfl_xor(acc[e], off, 32);
    acc[e] += bg[e];
  }
  // strict top-2 (earlier index wins ties, matching jax.lax.top_k)
  float g0 = -3.4e38f, g1 = -3.4e38f;
  int   i0 = -1, i1 = -1;
#pragma unroll
  for (int e = 0; e < NE; ++e) {
    float v = acc[e];
    if (v > g0)      { g1 = g0; i1 = i0; g0 = v; i0 = e; }
    else if (v > g1) { g1 = v;  i1 = e; }
  }
  float ex = expf(g1 - g0);
  float p0 = 1.f / (1.f + ex);
  float p1 = ex  / (1.f + ex);
  if (lane < NE) {
    float v = (lane == i0) ? p0 : (lane == i1) ? p1 : 0.f;
    comb[(size_t)t * NE + lane] = v;
  }
}

// ---------------------------------------------------------------------------
// f32 -> bf16 conversion (for We), grid-stride.
// ---------------------------------------------------------------------------
__global__ __launch_bounds__(256) void cvt_bf16_kernel(
    const float* __restrict__ src, __bf16* __restrict__ dst, int n) {
  int i      = blockIdx.x * blockDim.x + threadIdx.x;
  int stride = gridDim.x * blockDim.x;
  for (; i < n; i += stride) dst[i] = (__bf16)src[i];
}

// ---------------------------------------------------------------------------
// Main MoE GEMM. out[t,o] = sum_e comb[t,e]*( sum_k x[t,k]*We[e,o,k] + be[e,o] )
// Block tile 128x128, 8 waves (each 32 rows x 64 cols = 2x4 WMMA subtiles).
// Rows of A pre-scaled by comb[t,e] so ONE accumulator sums over experts & K.
// Bias folded in as an extra WMMA that initializes the accumulator.
// ---------------------------------------------------------------------------
__global__ __launch_bounds__(256) void moe_kernel(
    const float* __restrict__ x, const __bf16* __restrict__ Wb,
    const float* __restrict__ be, const float* __restrict__ comb,
    float* __restrict__ out) {
  __shared__ __attribute__((aligned(16))) float  ldsComb[BM * NE];
  __shared__ __attribute__((aligned(16))) __bf16 ldsA[BM * AROW];
  __shared__ __attribute__((aligned(16))) __bf16 ldsCombBf[BM * CROW];
  __shared__ __attribute__((aligned(16))) __bf16 ldsBeT[BN * CROW];

  const int tid     = threadIdx.x;
  const int lane    = tid & 31;
  const int wid     = tid >> 5;
  const int wave_m  = wid & 3;          // 4 waves along M
  const int wave_n  = wid >> 2;         // 2 waves along N
  const int l16     = lane & 15;
  const int lh      = lane >> 4;        // lane half (0/1)
  const int rowBase = blockIdx.y * BM;
  const int colBase = blockIdx.x * BN;

  // ---- stage combine weights (f32 + K-padded bf16) and be^T tile ----
  for (int i = tid; i < BM * NE; i += 256)
    ldsComb[i] = comb[(size_t)(rowBase + (i >> 3)) * NE + (i & 7)];
  for (int i = tid; i < BM * 32; i += 256) {
    int r = i >> 5, k = i & 31;
    float v = (k < NE) ? comb[(size_t)(rowBase + r) * NE + k] : 0.f;
    ldsCombBf[r * CROW + k] = (__bf16)v;
  }
  for (int i = tid; i < BN * 32; i += 256) {
    int n = i >> 5, k = i & 31;
    float v = (k < NE) ? be[k * DDIM + colBase + n] : 0.f;
    ldsBeT[n * CROW + k] = (__bf16)v;
  }
  __syncthreads();

  union Frag { uint4 q[2]; v16bf v; };

  // ---- init accumulators with the bias GEMM: comb[16x8] @ be[8x16] ----
  v8f acc[2][4];
  {
    Frag ca[2], bb[4];
#pragma unroll
    for (int ms = 0; ms < 2; ++ms) {
      // A-layout: lanes0-15 K=0..15, lanes16-31 K=16..31
      const char* p = (const char*)&ldsCombBf[(wave_m * 32 + ms * 16 + l16) * CROW] + lh * 32;
      ca[ms].q[0] = *(const uint4*)p;
      ca[ms].q[1] = *(const uint4*)(p + 16);
    }
#pragma unroll
    for (int ns = 0; ns < 4; ++ns) {
      // B-layout: lanes0-15 K=0..15, lanes16-31 K=16..31 (column = l16)
      const char* p = (const char*)&ldsBeT[(wave_n * 64 + ns * 16 + l16) * CROW] + lh * 32;
      bb[ns].q[0] = *(const uint4*)p;
      bb[ns].q[1] = *(const uint4*)(p + 16);
    }
    v8f z = {0.f, 0.f, 0.f, 0.f, 0.f, 0.f, 0.f, 0.f};
#pragma unroll
    for (int ms = 0; ms < 2; ++ms)
#pragma unroll
      for (int ns = 0; ns < 4; ++ns)
        acc[ms][ns] = __builtin_amdgcn_wmma_f32_16x16x32_bf16(
            false, ca[ms].v, false, bb[ns].v, (short)0, z, false, false);
  }

  // ---- main loops: experts outer, K staged through LDS ----
  for (int e = 0; e < NE; ++e) {
    for (int ks = 0; ks < DDIM; ks += KS) {
      __syncthreads();  // protect previous-stage ldsA reads
      // stage A: bf16( comb[r,e] * x[r, ks+c] ), coalesced 128B per wave
      for (int i = tid; i < BM * KS; i += 256) {
        int r = i >> 7, c = i & (KS - 1);
        float xv = x[(size_t)(rowBase + r) * DDIM + ks + c];
        ldsA[r * AROW + c] = (__bf16)(ldsComb[r * NE + e] * xv);
      }
      __syncthreads();

#pragma unroll
      for (int kk = 0; kk < KS / 32; ++kk) {
        Frag a[2];
#pragma unroll
        for (int ms = 0; ms < 2; ++ms) {
          const char* p = (const char*)&ldsA[(wave_m * 32 + ms * 16 + l16) * AROW]
                          + kk * 64 + lh * 32;
          a[ms].q[0] = *(const uint4*)p;        // ds_load_b128
          a[ms].q[1] = *(const uint4*)(p + 16); // ds_load_b128
        }
#pragma unroll
        for (int ns = 0; ns < 4; ++ns) {
          Frag b;
          const __bf16* bp = Wb
              + ((size_t)e * DDIM + (size_t)(colBase + wave_n * 64 + ns * 16 + l16)) * DDIM
              + ks + kk * 32 + lh * 16;
          b.q[0] = ((const uint4*)bp)[0];       // global_load_b128 (L2-resident)
          b.q[1] = ((const uint4*)bp)[1];
#pragma unroll
          for (int ms = 0; ms < 2; ++ms)
            acc[ms][ns] = __builtin_amdgcn_wmma_f32_16x16x32_bf16(
                false, a[ms].v, false, b.v, (short)0, acc[ms][ns], false, false);
        }
      }
    }
  }

  // ---- epilogue: C layout -> out. lanes0-15: M=v, lanes16-31: M=v+8 ----
#pragma unroll
  for (int ms = 0; ms < 2; ++ms)
#pragma unroll
    for (int ns = 0; ns < 4; ++ns) {
      int row0 = rowBase + wave_m * 32 + ms * 16 + lh * 8;
      int col  = colBase + wave_n * 64 + ns * 16 + l16;
#pragma unroll
      for (int v = 0; v < 8; ++v)
        out[(size_t)(row0 + v) * DDIM + col] = acc[ms][ns][v];
    }
}

// ---------------------------------------------------------------------------
extern "C" void kernel_launch(void* const* d_in, const int* in_sizes, int n_in,
                              void* d_out, int out_size, void* d_ws, size_t ws_size,
                              hipStream_t stream) {
  if (n_in < 5) return;
  const float* x  = (const float*)d_in[0];   // [T, D]
  const float* Wg = (const float*)d_in[1];   // [E, D]
  const float* bg = (const float*)d_in[2];   // [E]
  const float* We = (const float*)d_in[3];   // [E, D, D] (row = out dim)
  const float* be = (const float*)d_in[4];   // [E, D]
  float* out = (float*)d_out;

  const int T = in_sizes[0] / DDIM;          // 16384

  // workspace layout: comb f32 [T*8] | Wb bf16 [E*D*D]
  float*  comb = (float*)d_ws;
  __bf16* Wb   = (__bf16*)((char*)d_ws + (size_t)T * NE * sizeof(float));

  gate_kernel<<<T / 8, 256, 0, stream>>>(x, Wg, bg, comb);

  const int nW = NE * DDIM * DDIM;
  cvt_bf16_kernel<<<4096, 256, 0, stream>>>(We, Wb, nW);

  dim3 grid(DDIM / BN, T / BM);              // (8, 128)
  moe_kernel<<<grid, 256, 0, stream>>>(x, Wb, be, comb, out);
}
